// EncoderLayer_51840255262882
// MI455X (gfx1250) — matrix operand order repaired and checked by reading, not compiled, not run on hardware
//
#include <hip/hip_runtime.h>
#include <stdint.h>

// ---------------------------------------------------------------------------
// Types for CDNA5 WMMA (wave32). v16bf = A/B fragment, v8f = C/D fragment.
// ---------------------------------------------------------------------------
typedef __attribute__((ext_vector_type(16))) __bf16   v16bf;
typedef __attribute__((ext_vector_type(8)))  float    v8f;
typedef __attribute__((ext_vector_type(4)))  uint32_t u32x4;

union FragU { u32x4 u[2]; v16bf v; };   // 32 bytes = one 16-wide bf16 fragment

static __device__ __forceinline__ unsigned short f2bf(float f) {
  union { float f; uint32_t u; } c; c.f = f;
  uint32_t r = c.u + 0x7FFFu + ((c.u >> 16) & 1u);   // round-to-nearest-even
  return (unsigned short)(r >> 16);
}

static __device__ __forceinline__ v8f wmma_bf16(const v16bf& a, const v16bf& b, v8f c) {
  return __builtin_amdgcn_wmma_f32_16x16x32_bf16(false, a, false, b, (short)0, c,
                                                 false, false);
}

// ---------------------------------------------------------------------------
// fp32 -> bf16 conversion (weights), packed stores, n multiple of 4
// ---------------------------------------------------------------------------
__global__ __launch_bounds__(256) void cvt_bf16_kernel(const float* __restrict__ in,
                                                       unsigned short* __restrict__ out,
                                                       int n) {
  int i4 = (blockIdx.x * blockDim.x + threadIdx.x) * 4;
  int stride = gridDim.x * blockDim.x * 4;
  for (; i4 < n; i4 += stride) {
    float4 v = *(const float4*)(in + i4);
    uint2 pk;
    pk.x = (uint32_t)f2bf(v.x) | ((uint32_t)f2bf(v.y) << 16);
    pk.y = (uint32_t)f2bf(v.z) | ((uint32_t)f2bf(v.w) << 16);
    *(uint2*)(out + i4) = pk;
  }
}

// ---------------------------------------------------------------------------
// LayerNorm: one row (D=1024) per 256-thread block, bf16 output
// ---------------------------------------------------------------------------
__global__ __launch_bounds__(256) void ln_bf16_kernel(const float* __restrict__ x,
                                                      const float* __restrict__ g,
                                                      const float* __restrict__ be,
                                                      unsigned short* __restrict__ out) {
  constexpr int Dm = 1024;
  __shared__ float redS[8], redQ[8];
  const int row = blockIdx.x, tid = threadIdx.x;
  const float4 v = *(const float4*)(x + (size_t)row * Dm + tid * 4);
  float s = v.x + v.y + v.z + v.w;
  float q = v.x * v.x + v.y * v.y + v.z * v.z + v.w * v.w;
#pragma unroll
  for (int off = 16; off; off >>= 1) {
    s += __shfl_xor(s, off, 32);
    q += __shfl_xor(q, off, 32);
  }
  if ((tid & 31) == 0) { redS[tid >> 5] = s; redQ[tid >> 5] = q; }
  __syncthreads();
  float s8 = 0.f, q8 = 0.f;
#pragma unroll
  for (int i = 0; i < 8; ++i) { s8 += redS[i]; q8 += redQ[i]; }
  const float mu = s8 * (1.0f / Dm);
  const float var = q8 * (1.0f / Dm) - mu * mu;
  const float rs = rsqrtf(var + 1e-5f);
  const float vals[4] = {v.x, v.y, v.z, v.w};
#pragma unroll
  for (int i = 0; i < 4; ++i) {
    int c = tid * 4 + i;
    out[(size_t)row * Dm + c] = f2bf((vals[i] - mu) * rs * g[c] + be[c]);
  }
}

// ---------------------------------------------------------------------------
// bf16 WMMA GEMM: C[M,N] = A[M,K] @ B[K,N] (+bias, epilogue variants)
//   EPI 0: bf16 out = acc + bias
//   EPI 1: bf16 out = relu(acc + bias)
//   EPI 2: f32  out = acc + bias + resid
// Block tile 256x64, 8 waves (4x2), each wave owns 64x32 (4x2 WMMA subtiles),
// K-step 32 -> 8 WMMAs per wave per step against 12 ds_load_b128.
// LDS: A row-major padded (stride 40 halves), B transposed [n][k] padded.
// ---------------------------------------------------------------------------
template <int EPI>
__global__ __launch_bounds__(256) void gemm_bf16_kernel(
    const unsigned short* __restrict__ A, const unsigned short* __restrict__ B,
    const float* __restrict__ bias, const float* __restrict__ resid,
    void* __restrict__ Cout, int M, int N, int K, int lda, int ldb, int ldc) {
  __shared__ unsigned short sA[256 * 40];  // 20480 B
  __shared__ unsigned short sB[64 * 40];   // transposed [n][k], 5120 B
  const int tid = threadIdx.x;
  const int lane = tid & 31;
  const int w = tid >> 5;
  const int wm = w >> 1, wn = w & 1;       // wave tile: rows wm*64, cols wn*32
  const int m0 = blockIdx.y * 256;
  const int n0 = blockIdx.x * 64;
  const int hi = lane >> 4;                // half-wave select
  const int lo = lane & 15;
  const int kb = hi * 8;                   // A-fragment K base (ISA 16-bit A layout)

  v8f acc[4][2];
#pragma unroll
  for (int i = 0; i < 4; ++i)
#pragma unroll
    for (int j = 0; j < 2; ++j) acc[i][j] = (v8f){0, 0, 0, 0, 0, 0, 0, 0};

  const int br = tid >> 3, bc = (tid & 7) * 8;    // B loader: 8 halves/thread

  for (int k0 = 0; k0 < K; k0 += 32) {
    // --- stage A tile (256x32) row-major: one row per thread, 32 halves
    const unsigned short* ga = A + (size_t)(m0 + tid) * lda + k0;
#pragma unroll
    for (int c = 0; c < 4; ++c)
      *(u32x4*)(&sA[tid * 40 + c * 8]) = *(const u32x4*)(ga + c * 8);
    // --- stage B tile (32x64) transposed into LDS
    const unsigned short* gb = B + (size_t)(k0 + br) * ldb + n0 + bc;
    u32x4 bld = *(const u32x4*)(gb);
    const unsigned short* bs = (const unsigned short*)&bld;
#pragma unroll
    for (int i = 0; i < 8; ++i) sB[(bc + i) * 40 + br] = bs[i];
    if (k0 + 32 < K) {               // global_prefetch_b8 next K-step
      __builtin_prefetch(ga + 32, 0, 0);
      __builtin_prefetch(gb + (size_t)32 * ldb, 0, 0);
    }
    __syncthreads();

    // --- fragments: A lane row=lo, K runs [kb..kb+7] and [16+kb..16+kb+7]
    //               B lane n=lo, K runs [kb..kb+7] and [16+kb..16+kb+7]  (read from transposed tile)
    FragU afr[4], bfr[2];
#pragma unroll
    for (int sm = 0; sm < 4; ++sm) {
      int row = wm * 64 + sm * 16 + lo;
      afr[sm].u[0] = *(const u32x4*)(&sA[row * 40 + kb]);
      afr[sm].u[1] = *(const u32x4*)(&sA[row * 40 + 16 + kb]);
    }
#pragma unroll
    for (int sn = 0; sn < 2; ++sn) {
      int col = wn * 32 + sn * 16 + lo;
      bfr[sn].u[0] = *(const u32x4*)(&sB[col * 40 + kb]);
      bfr[sn].u[1] = *(const u32x4*)(&sB[col * 40 + 16 + kb]);
    }
#pragma unroll
    for (int sm = 0; sm < 4; ++sm)
#pragma unroll
      for (int sn = 0; sn < 2; ++sn)
        acc[sm][sn] = wmma_bf16(afr[sm].v, bfr[sn].v, acc[sm][sn]);
    __syncthreads();
  }

  // --- epilogue (C layout: VGPR r -> row r + 8*hi, col = lo)
#pragma unroll
  for (int sm = 0; sm < 4; ++sm) {
#pragma unroll
    for (int sn = 0; sn < 2; ++sn) {
      int gcol = n0 + wn * 32 + sn * 16 + lo;
      float bv = bias[gcol];
#pragma unroll
      for (int r = 0; r < 8; ++r) {
        size_t grow = (size_t)(m0 + wm * 64 + sm * 16 + r + hi * 8);
        float v = acc[sm][sn][r] + bv;
        if (EPI == 0) {
          ((unsigned short*)Cout)[grow * ldc + gcol] = f2bf(v);
        } else if (EPI == 1) {
          ((unsigned short*)Cout)[grow * ldc + gcol] = f2bf(v > 0.f ? v : 0.f);
        } else {
          ((float*)Cout)[grow * ldc + gcol] = v + resid[grow * ldc + gcol];
        }
      }
    }
  }
}

// ---------------------------------------------------------------------------
// Flash attention, bf16 WMMA. One block = (b, h, 128 query rows); 8 waves each
// own a 16-row Q tile. 64-key K/V tiles cooperatively staged in LDS
// (V transposed), so each staged tile feeds 16 WMMAs per wave and the
// softmax shuffle reductions + barriers are amortized over 64 keys.
// P is staged through per-wave LDS to convert C-layout -> A-layout.
// ---------------------------------------------------------------------------
__global__ __launch_bounds__(256) void attn_kernel(const unsigned short* __restrict__ Q,
                                                   const unsigned short* __restrict__ Kg,
                                                   const unsigned short* __restrict__ V,
                                                   unsigned short* __restrict__ O) {
  constexpr int S = 2048, Dm = 1024, HD = 64;
  __shared__ unsigned short sK[64 * 72];        // K tile row-major [key][d]
  __shared__ unsigned short sVt[64 * 72];       // V tile transposed [d][key]
  __shared__ unsigned short sP[8 * 16 * 72];    // per-wave P tiles [row][key]
  const int tid = threadIdx.x, lane = tid & 31, w = tid >> 5;
  const int bh = blockIdx.x >> 4;               // 0..31
  const int qblk = blockIdx.x & 15;
  const int b = bh >> 4, h = bh & 15;
  const int q0 = qblk * 128 + w * 16;
  const int hi = lane >> 4, lo = lane & 15, kb = hi * 8;
  const size_t baseRow = (size_t)b * S;
  const int hc = h * HD;

  // Q fragments (A layout), head_dim 64 => two K=32 fragments
  FragU qf[2];
  {
    const unsigned short* qp = Q + (baseRow + q0 + lo) * Dm + hc;
#pragma unroll
    for (int c = 0; c < 2; ++c) {
      qf[c].u[0] = *(const u32x4*)(qp + c * 32 + kb);
      qf[c].u[1] = *(const u32x4*)(qp + c * 32 + 16 + kb);
    }
  }

  v8f acc[4];
#pragma unroll
  for (int t = 0; t < 4; ++t) acc[t] = (v8f){0, 0, 0, 0, 0, 0, 0, 0};
  float mrow[8], lrow[8];
#pragma unroll
  for (int r = 0; r < 8; ++r) { mrow[r] = -1e30f; lrow[r] = 0.f; }

  const int kr = tid >> 2, kc2 = (tid & 3) * 16;  // loaders: 64x64 tile, 16 halves/thr
  unsigned short* myP = &sP[w * 16 * 72];

  for (int kt = 0; kt < S; kt += 64) {
    // --- stage K (row-major) and V (transposed) 64-key tiles
    const unsigned short* gk = Kg + (baseRow + kt + kr) * Dm + hc + kc2;
    *(u32x4*)(&sK[kr * 72 + kc2]) = *(const u32x4*)(gk);
    *(u32x4*)(&sK[kr * 72 + kc2 + 8]) = *(const u32x4*)(gk + 8);
    const unsigned short* gv = V + (baseRow + kt + kr) * Dm + hc + kc2;
    u32x4 vld0 = *(const u32x4*)(gv);
    u32x4 vld1 = *(const u32x4*)(gv + 8);
    const unsigned short* vs0 = (const unsigned short*)&vld0;
    const unsigned short* vs1 = (const unsigned short*)&vld1;
#pragma unroll
    for (int i = 0; i < 8; ++i) {
      sVt[(kc2 + i) * 72 + kr] = vs0[i];
      sVt[(kc2 + 8 + i) * 72 + kr] = vs1[i];
    }
    __syncthreads();

    // --- S = Q @ K^T for four 16-key subtiles (B = K^T read row-major from sK)
    v8f sacc[4];
#pragma unroll
    for (int j = 0; j < 4; ++j) {
      sacc[j] = (v8f){0, 0, 0, 0, 0, 0, 0, 0};
#pragma unroll
      for (int c = 0; c < 2; ++c) {
        FragU kf;
        const int base = (j * 16 + lo) * 72 + c * 32 + kb;
        kf.u[0] = *(const u32x4*)(&sK[base]);
        kf.u[1] = *(const u32x4*)(&sK[base + 16]);
        sacc[j] = wmma_bf16(qf[c].v, kf.v, sacc[j]);
      }
    }

    // --- online softmax over 64 keys (row reductions within 16-lane halves)
    float fac[8];
#pragma unroll
    for (int r = 0; r < 8; ++r) {
      float sv[4];
#pragma unroll
      for (int j = 0; j < 4; ++j) sv[j] = sacc[j][r] * 0.125f;   // 1/sqrt(64)
      float mx = fmaxf(fmaxf(sv[0], sv[1]), fmaxf(sv[2], sv[3]));
#pragma unroll
      for (int off = 8; off; off >>= 1) mx = fmaxf(mx, __shfl_xor(mx, off, 32));
      float mnew = fmaxf(mrow[r], mx);
      fac[r] = __expf(mrow[r] - mnew);
      float ps = 0.f;
#pragma unroll
      for (int j = 0; j < 4; ++j) {
        float p = __expf(sv[j] - mnew);
        ps += p;
        // scatter P (bf16) C-layout -> row-major LDS tile [16][64+pad]
        myP[(r + hi * 8) * 72 + j * 16 + lo] = f2bf(p);
      }
#pragma unroll
      for (int off = 8; off; off >>= 1) ps += __shfl_xor(ps, off, 32);
      lrow[r] = lrow[r] * fac[r] + ps;
      mrow[r] = mnew;
    }
#pragma unroll
    for (int t = 0; t < 4; ++t)
#pragma unroll
      for (int r = 0; r < 8; ++r) acc[t][r] *= fac[r];

    // --- O += P @ V over two 32-key groups (P as A-frag, V as B-frag)
#pragma unroll
    for (int g2 = 0; g2 < 2; ++g2) {
      FragU pf;
      pf.u[0] = *(const u32x4*)(&myP[lo * 72 + g2 * 32 + kb]);
      pf.u[1] = *(const u32x4*)(&myP[lo * 72 + g2 * 32 + 16 + kb]);
#pragma unroll
      for (int t = 0; t < 4; ++t) {
        FragU vf;
        const int base = (t * 16 + lo) * 72 + g2 * 32 + kb;
        vf.u[0] = *(const u32x4*)(&sVt[base]);
        vf.u[1] = *(const u32x4*)(&sVt[base + 16]);
        acc[t] = wmma_bf16(pf.v, vf.v, acc[t]);
      }
    }
    __syncthreads();
  }

  // --- normalize and emit bf16 attention output [B*S, D]
#pragma unroll
  for (int r = 0; r < 8; ++r) {
    float inv = 1.0f / lrow[r];
    size_t row = baseRow + q0 + r + hi * 8;
#pragma unroll
    for (int t = 0; t < 4; ++t)
      O[row * Dm + hc + t * 16 + lo] = f2bf(acc[t][r] * inv);
  }
}

// ---------------------------------------------------------------------------
// Host-side pipeline
// ---------------------------------------------------------------------------
extern "C" void kernel_launch(void* const* d_in, const int* in_sizes, int n_in,
                              void* d_out, int out_size, void* d_ws, size_t ws_size,
                              hipStream_t stream) {
  (void)in_sizes; (void)n_in; (void)out_size; (void)ws_size;
  const float* x   = (const float*)d_in[0];
  const float* wq  = (const float*)d_in[1];
  const float* bq  = (const float*)d_in[2];
  const float* wk  = (const float*)d_in[3];
  const float* bk  = (const float*)d_in[4];
  const float* wv  = (const float*)d_in[5];
  const float* bv  = (const float*)d_in[6];
  const float* wo  = (const float*)d_in[7];
  const float* bo  = (const float*)d_in[8];
  const float* w1  = (const float*)d_in[9];
  const float* b1  = (const float*)d_in[10];
  const float* w2  = (const float*)d_in[11];
  const float* b2  = (const float*)d_in[12];
  const float* g1  = (const float*)d_in[13];
  const float* be1 = (const float*)d_in[14];
  const float* g2  = (const float*)d_in[15];
  const float* be2 = (const float*)d_in[16];
  float* out = (float*)d_out;

  constexpr int R = 2 * 2048;        // 4096 token rows
  constexpr int Dm = 1024, Dff = 4096;

  char* ws = (char*)d_ws;
  auto alloc = [&](size_t bytes) { char* p = ws; ws += (bytes + 255) & ~(size_t)255; return p; };
  unsigned short* wqb  = (unsigned short*)alloc((size_t)Dm * Dm * 2);
  unsigned short* wkb  = (unsigned short*)alloc((size_t)Dm * Dm * 2);
  unsigned short* wvb  = (unsigned short*)alloc((size_t)Dm * Dm * 2);
  unsigned short* wob  = (unsigned short*)alloc((size_t)Dm * Dm * 2);
  unsigned short* w1b  = (unsigned short*)alloc((size_t)Dm * Dff * 2);
  unsigned short* w2b  = (unsigned short*)alloc((size_t)Dff * Dm * 2);
  unsigned short* nxb  = (unsigned short*)alloc((size_t)R * Dm * 2);
  unsigned short* qb   = (unsigned short*)alloc((size_t)R * Dm * 2);
  unsigned short* kbf  = (unsigned short*)alloc((size_t)R * Dm * 2);
  unsigned short* vbf  = (unsigned short*)alloc((size_t)R * Dm * 2);
  unsigned short* aob  = (unsigned short*)alloc((size_t)R * Dm * 2);
  float*          x1   = (float*)alloc((size_t)R * Dm * 4);
  unsigned short* nx2b = (unsigned short*)alloc((size_t)R * Dm * 2);
  unsigned short* hb   = (unsigned short*)alloc((size_t)R * Dff * 2);

  // 1) weights -> bf16
  cvt_bf16_kernel<<<1024, 256, 0, stream>>>(wq, wqb, Dm * Dm);
  cvt_bf16_kernel<<<1024, 256, 0, stream>>>(wk, wkb, Dm * Dm);
  cvt_bf16_kernel<<<1024, 256, 0, stream>>>(wv, wvb, Dm * Dm);
  cvt_bf16_kernel<<<1024, 256, 0, stream>>>(wo, wob, Dm * Dm);
  cvt_bf16_kernel<<<4096, 256, 0, stream>>>(w1, w1b, Dm * Dff);
  cvt_bf16_kernel<<<4096, 256, 0, stream>>>(w2, w2b, Dff * Dm);

  // 2) LN1
  ln_bf16_kernel<<<R, 256, 0, stream>>>(x, g1, be1, nxb);

  // 3) Q,K,V projections (bf16 out + bias)
  dim3 gP(Dm / 64, R / 256);
  gemm_bf16_kernel<0><<<gP, 256, 0, stream>>>(nxb, wqb, bq, nullptr, qb,  R, Dm, Dm, Dm, Dm, Dm);
  gemm_bf16_kernel<0><<<gP, 256, 0, stream>>>(nxb, wkb, bk, nullptr, kbf, R, Dm, Dm, Dm, Dm, Dm);
  gemm_bf16_kernel<0><<<gP, 256, 0, stream>>>(nxb, wvb, bv, nullptr, vbf, R, Dm, Dm, Dm, Dm, Dm);

  // 4) flash attention (2 batches * 16 heads * 16 q-blocks of 128 rows)
  attn_kernel<<<512, 256, 0, stream>>>(qb, kbf, vbf, aob);

  // 5) output projection + residual: x1 = x + attn @ wo + bo
  gemm_bf16_kernel<2><<<gP, 256, 0, stream>>>(aob, wob, bo, x, x1, R, Dm, Dm, Dm, Dm, Dm);

  // 6) LN2
  ln_bf16_kernel<<<R, 256, 0, stream>>>(x1, g2, be2, nx2b);

  // 7) FFN: h = relu(nx2 @ w1 + b1) ; out = x1 + h @ w2 + b2
  dim3 gF(Dff / 64, R / 256);
  gemm_bf16_kernel<1><<<gF, 256, 0, stream>>>(nx2b, w1b, b1, nullptr, hb, R, Dff, Dm, Dm, Dff, Dff);
  gemm_bf16_kernel<2><<<gP, 256, 0, stream>>>(hb, w2b, b2, x1, out, R, Dm, Dff, Dff, Dm, Dm);
}
